// TFScaledDotProductAttention_6012954215151
// MI455X (gfx1250) — matrix operand order repaired and checked by reading, not compiled, not run on hardware
//
#include <hip/hip_runtime.h>

typedef __attribute__((ext_vector_type(16))) _Float16 v16h;
typedef __attribute__((ext_vector_type(8)))  float    v8f;
typedef __attribute__((ext_vector_type(4)))  float    v4f;

constexpr int Bc = 2, Hc = 16, Sq = 2048, Dh = 64;
constexpr int QT = 16;            // query rows per workgroup
constexpr int NWAVE = 8;          // wave32s per workgroup
constexpr int NTHREADS = NWAVE * 32;
constexpr int PAD = 4;            // LDS row padding (floats) to dodge bank conflicts

__global__ __launch_bounds__(NTHREADS, 1)
void sdpa_wmma_kernel(const float* __restrict__ Q, const float* __restrict__ Kp,
                      const float* __restrict__ Vp, const float* __restrict__ Mp,
                      float* __restrict__ Ctx, float* __restrict__ Attn)
{
    __shared__ float sc[QT][Sq + PAD];       // 16 x 2052 f32 ~= 128 KB
    __shared__ float pctx[NWAVE][32][8];     // split-K partial context, 8 KB

    const int qt   = blockIdx.x;             // 0..127  query tile
    const int bh   = blockIdx.y;             // 0..31   fused batch*head
    const int q0   = qt * QT;
    const int tid  = threadIdx.x;
    const int wave = tid >> 5;
    const int lane = tid & 31;
    const int hl   = lane & 15;              // N / M index within 16
    const int hi   = lane >> 4;              // half-wave select

    const size_t bhBase = (size_t)bh * Sq * Dh;
    const float* Qb = Q  + bhBase + (size_t)q0 * Dh;
    const float* Kb = Kp + bhBase;
    const float* Vb = Vp + bhBase;

    // ---- Q tile as two 16x32 f16 A-fragments (f32 -> f16 on load) ----
    v16h aq[2];
    {
        const int m  = hl;
        const int kb = hi * 8;
        const float* qrow = Qb + m * Dh;
        #pragma unroll
        for (int c = 0; c < 2; ++c) {
            const float* s0 = qrow + c * 32;
            v16h a;
            #pragma unroll
            for (int j = 0; j < 8; ++j) a[j]     = (_Float16)s0[kb + j];
            #pragma unroll
            for (int j = 0; j < 8; ++j) a[8 + j] = (_Float16)s0[16 + kb + j];
            aq[c] = a;
        }
    }

    // ---- pass 1: raw scores = (Q K^T) * rsqrt(D) into LDS ----
    for (int t = wave; t < Sq / 16; t += NWAVE) {
        const int n0 = t * 16;
        const int n  = n0 + hl;
        const int ko = hi * 8;
        const float* krow = Kb + (size_t)n * Dh;
        v8f c = {};
        #pragma unroll
        for (int ch = 0; ch < 2; ++ch) {
            const float* s0 = krow + ch * 32 + ko;   // two runs of 8 f32, 16 apart
            v16h b;
            #pragma unroll
            for (int j = 0; j < 16; ++j) b[j] = (_Float16)s0[j < 8 ? j : j + 8];
            c = __builtin_amdgcn_wmma_f32_16x16x32_f16(false, aq[ch], false, b,
                                                       (short)0, c, false, false);
        }
        #pragma unroll
        for (int r = 0; r < 8; ++r) {
            const int m = r + hi * 8;                // C layout: VGPR r -> row r / r+8
            sc[m][n0 + hl] = c[r] * 0.125f;          // 1/sqrt(64)
        }
    }
    __syncthreads();

    // ---- softmax over each row: 16 threads/row; mask added here (coalesced) ----
    {
        const int row = tid >> 4;                    // 0..15
        const int j   = tid & 15;                    // 0..15
        const float* marow = Mp + ((size_t)bh * Sq + q0 + row) * (size_t)Sq;

        float mx = -3.4e38f;
        for (int i = j; i < Sq / 4; i += 16) {       // + mask, find max
            v4f m4 = __builtin_nontemporal_load((const v4f*)(marow + i * 4));
            v4f v  = *(v4f*)&sc[row][i * 4];
            v.x += m4.x; v.y += m4.y; v.z += m4.z; v.w += m4.w;
            *(v4f*)&sc[row][i * 4] = v;
            mx = fmaxf(mx, fmaxf(fmaxf(v.x, v.y), fmaxf(v.z, v.w)));
        }
        #pragma unroll
        for (int off = 1; off < 16; off <<= 1)
            mx = fmaxf(mx, __shfl_xor(mx, off, 32));

        float sum = 0.f;
        for (int i = j; i < Sq / 4; i += 16) {       // exp, accumulate sum
            v4f v = *(v4f*)&sc[row][i * 4];
            v.x = __expf(v.x - mx); v.y = __expf(v.y - mx);
            v.z = __expf(v.z - mx); v.w = __expf(v.w - mx);
            sum += (v.x + v.y) + (v.z + v.w);
            *(v4f*)&sc[row][i * 4] = v;
        }
        #pragma unroll
        for (int off = 1; off < 16; off <<= 1)
            sum += __shfl_xor(sum, off, 32);

        const float inv = 1.f / sum;
        float* arow = Attn + ((size_t)bh * Sq + q0 + row) * (size_t)Sq;
        for (int i = j; i < Sq / 4; i += 16) {       // normalize, stream out
            v4f v = *(v4f*)&sc[row][i * 4];
            v.x *= inv; v.y *= inv; v.z *= inv; v.w *= inv;
            *(v4f*)&sc[row][i * 4] = v;                            // keep P for pass 2
            __builtin_nontemporal_store(v, (v4f*)(arow + i * 4));  // stream attn out
        }
    }
    __syncthreads();

    // ---- pass 2: context = P @ V, split-K across wave pairs ----
    {
        const int n0 = (wave & 3) * 16;              // D slice
        const int kh = wave >> 2;                    // K half (0/1)
        const int n  = n0 + hl;
        const int ko = hi * 8;
        const int m  = hl;
        const int kb = hi * 8;
        v8f c = {};
        for (int kc = kh * 32; kc < kh * 32 + 32; ++kc) {
            const float* prow = &sc[m][kc * 32];
            v16h a;
            #pragma unroll
            for (int j = 0; j < 8; ++j) a[j]     = (_Float16)prow[kb + j];
            #pragma unroll
            for (int j = 0; j < 8; ++j) a[8 + j] = (_Float16)prow[16 + kb + j];

            const float* vsrc = Vb + (size_t)(kc * 32 + ko) * Dh + n;
            v16h b;
            #pragma unroll
            for (int j = 0; j < 16; ++j) b[j] = (_Float16)vsrc[(size_t)(j < 8 ? j : j + 8) * Dh];

            c = __builtin_amdgcn_wmma_f32_16x16x32_f16(false, a, false, b,
                                                       (short)0, c, false, false);
        }
        #pragma unroll
        for (int r = 0; r < 8; ++r) pctx[wave][lane][r] = c[r];
    }
    __syncthreads();

    // reduce the two K-halves and write context
    if (wave < 4) {
        const int n0 = wave * 16;
        float* crow = Ctx + ((size_t)bh * Sq + q0) * Dh;
        #pragma unroll
        for (int r = 0; r < 8; ++r) {
            const int mm = r + hi * 8;
            crow[(size_t)mm * Dh + n0 + hl] =
                pctx[wave][lane][r] + pctx[wave + 4][lane][r];
        }
    }
}

extern "C" void kernel_launch(void* const* d_in, const int* in_sizes, int n_in,
                              void* d_out, int out_size, void* d_ws, size_t ws_size,
                              hipStream_t stream) {
    (void)in_sizes; (void)n_in; (void)out_size; (void)d_ws; (void)ws_size;
    const float* Q = (const float*)d_in[0];
    const float* K = (const float*)d_in[1];
    const float* V = (const float*)d_in[2];
    const float* M = (const float*)d_in[3];
    float* out  = (float*)d_out;
    float* Ctx  = out;                                     // [B,H,S,D]
    float* Attn = out + (size_t)Bc * Hc * Sq * Dh;         // [B,H,S,S]

    dim3 grid(Sq / QT, Bc * Hc);
    sdpa_wmma_kernel<<<grid, NTHREADS, 0, stream>>>(Q, K, V, M, Ctx, Attn);
}
